// SigLIP2EncoderLayer_30279519437330
// MI455X (gfx1250) — matrix operand order repaired and checked by reading, not compiled, not run on hardware
//
#include <hip/hip_runtime.h>

#define E_    1152
#define INTER 4304
#define INTERP 4320
#define HEADS 16
#define HD    72
#define DP    96
#define DVP   80
#define B_    8
#define S_    1024
#define R_    (B_*S_)
#define SCALE_Q 0.11785113019775792f  // 72^-0.5

typedef unsigned short u16;
typedef unsigned int   u32;
typedef __attribute__((ext_vector_type(16))) __bf16 v16bf;
typedef __attribute__((ext_vector_type(8)))  float  v8f;

union BU { u32 u[8]; v16bf v; };

// A-matrix element offset for (vgpr v, lane-half h): pairs along K
#define KOFFA(v,h) ((h)*16 + ((v)<<1))

__device__ __forceinline__ u16 f2bf(float f) {
  u32 u = __float_as_uint(f);
  u += 0x7fffu + ((u >> 16) & 1u);      // round-to-nearest-even
  return (u16)(u >> 16);
}
__device__ __forceinline__ float bf2f(u16 h) { return __uint_as_float(((u32)h) << 16); }
__device__ __forceinline__ u32 pack2(float a, float b) {
  return (u32)f2bf(a) | ((u32)f2bf(b) << 16);
}
__device__ __forceinline__ u32 ld2(const u16* p) { return *(const u32*)p; }

// error-compensated bf16 product: (Ah+Al)(Bh+Bl) ~= AhBh + AhBl + AlBh
__device__ __forceinline__ v8f wmma3(const BU& ah, const BU& al,
                                     const BU& bh, const BU& bl, v8f c) {
  c = __builtin_amdgcn_wmma_f32_16x16x32_bf16(false, ah.v, false, bh.v, (short)0, c, false, false);
  c = __builtin_amdgcn_wmma_f32_16x16x32_bf16(false, ah.v, false, bl.v, (short)0, c, false, false);
  c = __builtin_amdgcn_wmma_f32_16x16x32_bf16(false, al.v, false, bh.v, (short)0, c, false, false);
  return c;
}

// One wave computes C[16 x NT*16] = A[16 x 32*ksteps] * B^T tile, hi/lo split.
// All NT tiles are unconditionally active -> no accumulator phi-copies.
template<int NT>
__device__ __forceinline__ void wave_gemm(
    const u16* __restrict__ Ah, const u16* __restrict__ Al, int lda,
    const u16* __restrict__ Bh, const u16* __restrict__ Bl, int ldb,
    int row0, int col0, int ksteps, v8f* acc)
{
  const int lane = threadIdx.x & 31;
  const int half = lane >> 4;
  const int l15  = lane & 15;
  const u16* arh = Ah + (size_t)(row0 + l15) * lda;
  const u16* arl = Al + (size_t)(row0 + l15) * lda;
  for (int ks = 0; ks < ksteps; ++ks) {
    const int kb = ks * 32;
    BU ah, al;
#pragma unroll
    for (int v = 0; v < 8; ++v) {
      int k = kb + KOFFA(v, half);
      ah.u[v] = ld2(arh + k);
      al.u[v] = ld2(arl + k);
    }
#pragma unroll
    for (int t = 0; t < NT; ++t) {
      const u16* brh = Bh + (size_t)(col0 + 16*t + l15) * ldb + kb;
      const u16* brl = Bl + (size_t)(col0 + 16*t + l15) * ldb + kb;
      BU bh, bl;
#pragma unroll
      for (int v = 0; v < 8; ++v) {
        int k = half * 16 + 2 * v;
        bh.u[v] = ld2(brh + k);
        bl.u[v] = ld2(brl + k);
      }
      acc[t] = wmma3(ah, al, bh, bl, acc[t]);
    }
  }
}

// ---------------- helper kernels ----------------

__global__ void k_conv_hilo(const float* __restrict__ src, u16* __restrict__ dh,
                            u16* __restrict__ dl, int n) {
  int i = blockIdx.x * 256 + threadIdx.x;
  if (i < n) {
    float f = src[i];
    u16 hb = f2bf(f);
    dh[i] = hb;
    dl[i] = f2bf(f - bf2f(hb));
  }
}

__global__ void k_conv_hilo_pad(const float* __restrict__ src, u16* __restrict__ dh,
                                u16* __restrict__ dl, int srcld, int dstld, int n) {
  int i = blockIdx.x * 256 + threadIdx.x;
  if (i < n) {
    int r = i / srcld, c = i - r * srcld;
    float f = src[i];
    u16 hb = f2bf(f);
    size_t o = (size_t)r * dstld + c;
    dh[o] = hb;
    dl[o] = f2bf(f - bf2f(hb));
  }
}

__global__ void k_zero_pad(u32* __restrict__ p, int ldw, int c0, int w, int total) {
  int i = blockIdx.x * 256 + threadIdx.x;
  if (i < total) {
    int r = i / w, c = i - r * w;
    p[(size_t)r * ldw + c0 + c] = 0u;
  }
}

// wave-per-row layernorm -> bf16 hi/lo output
__global__ void k_layernorm(const float* __restrict__ x, const float* __restrict__ g,
                            const float* __restrict__ bta, u16* __restrict__ oh,
                            u16* __restrict__ ol) {
  int row  = blockIdx.x * 8 + (threadIdx.x >> 5);
  int lane = threadIdx.x & 31;
  const float* xr = x + (size_t)row * E_;
  float v[36];
  float s = 0.f;
#pragma unroll
  for (int i = 0; i < 36; ++i) { v[i] = xr[lane + 32 * i]; s += v[i]; }
#pragma unroll
  for (int m = 1; m < 32; m <<= 1) s += __shfl_xor(s, m, 32);
  float mu = s * (1.f / 1152.f);
  float q = 0.f;
#pragma unroll
  for (int i = 0; i < 36; ++i) { float d = v[i] - mu; q += d * d; }
#pragma unroll
  for (int m = 1; m < 32; m <<= 1) q += __shfl_xor(q, m, 32);
  float inv = rsqrtf(q * (1.f / 1152.f) + 1e-6f);
#pragma unroll
  for (int i = 0; i < 36; ++i) {
    int c = lane + 32 * i;
    float y = (v[i] - mu) * inv * g[c] + bta[c];
    u16 hb = f2bf(y);
    size_t o = (size_t)row * E_ + c;
    oh[o] = hb;
    ol[o] = f2bf(y - bf2f(hb));
  }
}

// ---------------- GEMM kernels ----------------

__global__ void __launch_bounds__(32) k_gemm_qkv(
    const u16* __restrict__ xh, const u16* __restrict__ xl,
    const u16* __restrict__ wh, const u16* __restrict__ wl,
    const float* __restrict__ bias,
    u16* __restrict__ qph, u16* __restrict__ qpl,
    u16* __restrict__ kph, u16* __restrict__ kpl,
    u16* __restrict__ vth, u16* __restrict__ vtl)
{
  int row0 = blockIdx.x * 16, col0 = blockIdx.y * 64;
  v8f acc[4];
#pragma unroll
  for (int t = 0; t < 4; ++t)
#pragma unroll
    for (int r = 0; r < 8; ++r) acc[t][r] = 0.f;
  wave_gemm<4>(xh, xl, E_, wh, wl, E_, row0, col0, E_ / 32, acc);
  int lane = threadIdx.x & 31, half = lane >> 4, l15 = lane & 15;
#pragma unroll
  for (int t = 0; t < 4; ++t) {
    int nc  = col0 + 16 * t + l15;
    int sec = nc / E_;
    int ncc = nc - sec * E_;
    int h = ncc / HD, d = ncc - h * HD;
#pragma unroll
    for (int r = 0; r < 8; ++r) {
      int row = row0 + r + 8 * half;
      int b = row >> 10, sIdx = row & 1023;
      float c = acc[t][r] + bias[nc];
      if (sec == 0) {
        c *= SCALE_Q;
        size_t o = ((size_t)(b * HEADS + h) * S_ + sIdx) * DP + d;
        u16 hb = f2bf(c); qph[o] = hb; qpl[o] = f2bf(c - bf2f(hb));
      } else if (sec == 1) {
        size_t o = ((size_t)(b * HEADS + h) * S_ + sIdx) * DP + d;
        u16 hb = f2bf(c); kph[o] = hb; kpl[o] = f2bf(c - bf2f(hb));
      } else {
        size_t o = ((size_t)(b * HEADS + h) * DVP + d) * S_ + sIdx;  // V transposed
        u16 hb = f2bf(c); vth[o] = hb; vtl[o] = f2bf(c - bf2f(hb));
      }
    }
  }
}

__global__ void __launch_bounds__(32) k_gemm_out(
    const u16* __restrict__ ah, const u16* __restrict__ al,
    const u16* __restrict__ wh, const u16* __restrict__ wl,
    const float* __restrict__ bias, const float* __restrict__ resid,
    float* __restrict__ h1)
{
  int row0 = blockIdx.x * 16, col0 = blockIdx.y * 64;
  v8f acc[4];
#pragma unroll
  for (int t = 0; t < 4; ++t)
#pragma unroll
    for (int r = 0; r < 8; ++r) acc[t][r] = 0.f;
  wave_gemm<4>(ah, al, E_, wh, wl, E_, row0, col0, E_ / 32, acc);
  int lane = threadIdx.x & 31, half = lane >> 4, l15 = lane & 15;
#pragma unroll
  for (int t = 0; t < 4; ++t) {
    int nc = col0 + 16 * t + l15;
#pragma unroll
    for (int r = 0; r < 8; ++r) {
      int row = row0 + r + 8 * half;
      size_t o = (size_t)row * E_ + nc;
      h1[o] = resid[o] + acc[t][r] + bias[nc];
    }
  }
}

__device__ __forceinline__ float gelu_tanh(float x) {
  return 0.5f * x * (1.f + tanhf(0.7978845608028654f * (x + 0.044715f * x * x * x)));
}

// fc1 main: full 64-wide column blocks (covers cols [0, 4288))
__global__ void __launch_bounds__(32) k_gemm_fc1(
    const u16* __restrict__ ah, const u16* __restrict__ al,
    const u16* __restrict__ wh, const u16* __restrict__ wl,
    const float* __restrict__ bias, u16* __restrict__ gh, u16* __restrict__ gl)
{
  int row0 = blockIdx.x * 16, col0 = blockIdx.y * 64;
  v8f acc[4];
#pragma unroll
  for (int t = 0; t < 4; ++t)
#pragma unroll
    for (int r = 0; r < 8; ++r) acc[t][r] = 0.f;
  wave_gemm<4>(ah, al, E_, wh, wl, E_, row0, col0, E_ / 32, acc);
  int lane = threadIdx.x & 31, half = lane >> 4, l15 = lane & 15;
#pragma unroll
  for (int t = 0; t < 4; ++t) {
    int nc = col0 + 16 * t + l15;
#pragma unroll
    for (int r = 0; r < 8; ++r) {
      int row = row0 + r + 8 * half;
      float y = gelu_tanh(acc[t][r] + bias[nc]);
      size_t o = (size_t)row * INTERP + nc;
      u16 hb = f2bf(y); gh[o] = hb; gl[o] = f2bf(y - bf2f(hb));
    }
  }
}

// fc1 tail: single 16-wide column block at col0 = 4288
__global__ void __launch_bounds__(32) k_gemm_fc1_tail(
    const u16* __restrict__ ah, const u16* __restrict__ al,
    const u16* __restrict__ wh, const u16* __restrict__ wl,
    const float* __restrict__ bias, u16* __restrict__ gh, u16* __restrict__ gl)
{
  int row0 = blockIdx.x * 16;
  const int col0 = 4288;
  v8f acc[1];
#pragma unroll
  for (int r = 0; r < 8; ++r) acc[0][r] = 0.f;
  wave_gemm<1>(ah, al, E_, wh, wl, E_, row0, col0, E_ / 32, acc);
  int lane = threadIdx.x & 31, half = lane >> 4, l15 = lane & 15;
  int nc = col0 + l15;
#pragma unroll
  for (int r = 0; r < 8; ++r) {
    int row = row0 + r + 8 * half;
    float y = gelu_tanh(acc[0][r] + bias[nc]);
    size_t o = (size_t)row * INTERP + nc;
    u16 hb = f2bf(y); gh[o] = hb; gl[o] = f2bf(y - bf2f(hb));
  }
}

__global__ void __launch_bounds__(32) k_gemm_fc2(
    const u16* __restrict__ ah, const u16* __restrict__ al,
    const u16* __restrict__ wh, const u16* __restrict__ wl,
    const float* __restrict__ bias, const float* __restrict__ h1,
    float* __restrict__ out)
{
  int row0 = blockIdx.x * 16, col0 = blockIdx.y * 64;
  v8f acc[4];
#pragma unroll
  for (int t = 0; t < 4; ++t)
#pragma unroll
    for (int r = 0; r < 8; ++r) acc[t][r] = 0.f;
  wave_gemm<4>(ah, al, INTERP, wh, wl, INTERP, row0, col0, INTERP / 32, acc);
  int lane = threadIdx.x & 31, half = lane >> 4, l15 = lane & 15;
#pragma unroll
  for (int t = 0; t < 4; ++t) {
    int nc = col0 + 16 * t + l15;
#pragma unroll
    for (int r = 0; r < 8; ++r) {
      int row = row0 + r + 8 * half;
      size_t o = (size_t)row * E_ + nc;
      out[o] = h1[o] + acc[t][r] + bias[nc];
    }
  }
}

// ---------------- flash attention: one wave per (b,h,16 queries) ----------------

__global__ void __launch_bounds__(32) k_attn(
    const u16* __restrict__ qph, const u16* __restrict__ qpl,
    const u16* __restrict__ kph, const u16* __restrict__ kpl,
    const u16* __restrict__ vth, const u16* __restrict__ vtl,
    u16* __restrict__ oh, u16* __restrict__ ol)
{
  __shared__ float P[16][33];
  const int bh = blockIdx.x >> 6;
  const int q0 = (blockIdx.x & 63) << 4;
  const int lane = threadIdx.x & 31, half = lane >> 4, l15 = lane & 15;

  // Q A-tiles (3 k-steps of 32 over padded D=96), hi/lo, in registers
  const u16* qr_h = qph + ((size_t)bh * S_ + q0 + l15) * DP;
  const u16* qr_l = qpl + ((size_t)bh * S_ + q0 + l15) * DP;
  BU qa_h[3], qa_l[3];
#pragma unroll
  for (int ks = 0; ks < 3; ++ks)
#pragma unroll
    for (int v = 0; v < 8; ++v) {
      int k = ks * 32 + KOFFA(v, half);
      qa_h[ks].u[v] = ld2(qr_h + k);
      qa_l[ks].u[v] = ld2(qr_l + k);
    }

  v8f oacc[5];
#pragma unroll
  for (int nt = 0; nt < 5; ++nt)
#pragma unroll
    for (int r = 0; r < 8; ++r) oacc[nt][r] = 0.f;
  float mrow[8], lrow[8];
#pragma unroll
  for (int r = 0; r < 8; ++r) { mrow[r] = -1e30f; lrow[r] = 0.f; }

  const u16* kb_h = kph + (size_t)bh * S_ * DP;
  const u16* kb_l = kpl + (size_t)bh * S_ * DP;
  const u16* vb_h = vth + (size_t)bh * DVP * S_;
  const u16* vb_l = vtl + (size_t)bh * DVP * S_;

  for (int j = 0; j < 32; ++j) {            // kv blocks of 32
    const int kv0 = j * 32;
    v8f sc[2];
#pragma unroll
    for (int nt = 0; nt < 2; ++nt)
#pragma unroll
      for (int r = 0; r < 8; ++r) sc[nt][r] = 0.f;

    // scores: Q [16x96] * K^T [96x32]
#pragma unroll
    for (int nt = 0; nt < 2; ++nt) {
      const u16* kr_h = kb_h + (size_t)(kv0 + nt * 16 + l15) * DP;
      const u16* kr_l = kb_l + (size_t)(kv0 + nt * 16 + l15) * DP;
#pragma unroll
      for (int ks = 0; ks < 3; ++ks) {
        BU bh_, bl_;
#pragma unroll
        for (int v = 0; v < 8; ++v) {
          int k = ks * 32 + half * 16 + 2 * v;
          bh_.u[v] = ld2(kr_h + k);
          bl_.u[v] = ld2(kr_l + k);
        }
        sc[nt] = wmma3(qa_h[ks], qa_l[ks], bh_, bl_, sc[nt]);
      }
    }

    // online softmax update (row reductions across the 16 lanes of each half)
    float p0[8], p1[8], esc[8];
#pragma unroll
    for (int r = 0; r < 8; ++r) {
      float mx = fmaxf(sc[0][r], sc[1][r]);
#pragma unroll
      for (int m = 1; m < 16; m <<= 1) mx = fmaxf(mx, __shfl_xor(mx, m, 32));
      float mnew = fmaxf(mrow[r], mx);
      esc[r] = __expf(mrow[r] - mnew);
      p0[r] = __expf(sc[0][r] - mnew);
      p1[r] = __expf(sc[1][r] - mnew);
      float ps = p0[r] + p1[r];
#pragma unroll
      for (int m = 1; m < 16; m <<= 1) ps += __shfl_xor(ps, m, 32);
      lrow[r] = lrow[r] * esc[r] + ps;
      mrow[r] = mnew;
    }
#pragma unroll
    for (int nt = 0; nt < 5; ++nt)
#pragma unroll
      for (int r = 0; r < 8; ++r) oacc[nt][r] *= esc[r];

    // route P through LDS: C-layout -> A-layout (hi/lo bf16)
    __syncthreads();
#pragma unroll
    for (int r = 0; r < 8; ++r) {
      P[r + 8 * half][l15]      = p0[r];
      P[r + 8 * half][l15 + 16] = p1[r];
    }
    __syncthreads();
    BU pa_h, pa_l;
#pragma unroll
    for (int v = 0; v < 8; ++v) {
      int k = KOFFA(v, half);
      float f0 = P[l15][k], f1 = P[l15][k + 1];
      u16 h0 = f2bf(f0), h1b = f2bf(f1);
      pa_h.u[v] = (u32)h0 | ((u32)h1b << 16);
      pa_l.u[v] = pack2(f0 - bf2f(h0), f1 - bf2f(h1b));
    }

    // O += P [16x32] * V [32x80]  (V stored transposed)
#pragma unroll
    for (int nt = 0; nt < 5; ++nt) {
      const u16* vr_h = vb_h + (size_t)(nt * 16 + l15) * S_ + kv0;
      const u16* vr_l = vb_l + (size_t)(nt * 16 + l15) * S_ + kv0;
      BU bvh, bvl;
#pragma unroll
      for (int v = 0; v < 8; ++v) {
        int k = half * 16 + 2 * v;
        bvh.u[v] = ld2(vr_h + k);
        bvl.u[v] = ld2(vr_l + k);
      }
      oacc[nt] = wmma3(pa_h, pa_l, bvh, bvl, oacc[nt]);
    }
  }

  const int b = bh >> 4, h = bh & 15;
#pragma unroll
  for (int nt = 0; nt < 5; ++nt) {
    int d = nt * 16 + l15;
    if (d < HD) {
#pragma unroll
      for (int r = 0; r < 8; ++r) {
        int tok = (b << 10) + q0 + r + 8 * half;
        float y = oacc[nt][r] / lrow[r];
        size_t o = (size_t)tok * E_ + h * HD + d;
        u16 hb = f2bf(y);
        oh[o] = hb;
        ol[o] = f2bf(y - bf2f(hb));
      }
    }
  }
}

// ---------------- host launch ----------------

extern "C" void kernel_launch(void* const* d_in, const int* in_sizes, int n_in,
                              void* d_out, int out_size, void* d_ws, size_t ws_size,
                              hipStream_t stream) {
  (void)in_sizes; (void)n_in; (void)out_size; (void)ws_size;
  const float* hs   = (const float*)d_in[0];
  const float* ln1g = (const float*)d_in[1];
  const float* ln1b = (const float*)d_in[2];
  const float* ln2g = (const float*)d_in[3];
  const float* ln2b = (const float*)d_in[4];
  const float* qkvw = (const float*)d_in[5];
  const float* qkvb = (const float*)d_in[6];
  const float* outw = (const float*)d_in[7];
  const float* outb = (const float*)d_in[8];
  const float* fc1w = (const float*)d_in[9];
  const float* fc1b = (const float*)d_in[10];
  const float* fc2w = (const float*)d_in[11];
  const float* fc2b = (const float*)d_in[12];
  float* out = (float*)d_out;

  char* base = (char*)d_ws;
  size_t off = 0;
  auto alloc16 = [&](size_t elems) -> u16* {
    u16* r = (u16*)(base + off);
    off = (off + elems * sizeof(u16) + 255) & ~(size_t)255;
    return r;
  };
  u16 *xln_h  = alloc16((size_t)R_ * E_),      *xln_l  = alloc16((size_t)R_ * E_);
  u16 *qkvw_h = alloc16((size_t)3 * E_ * E_),  *qkvw_l = alloc16((size_t)3 * E_ * E_);
  u16 *outw_h = alloc16((size_t)E_ * E_),      *outw_l = alloc16((size_t)E_ * E_);
  u16 *fc1w_h = alloc16((size_t)INTER * E_),   *fc1w_l = alloc16((size_t)INTER * E_);
  u16 *fc2w_h = alloc16((size_t)E_ * INTERP),  *fc2w_l = alloc16((size_t)E_ * INTERP);
  u16 *qp_h = alloc16((size_t)B_ * HEADS * S_ * DP), *qp_l = alloc16((size_t)B_ * HEADS * S_ * DP);
  u16 *kp_h = alloc16((size_t)B_ * HEADS * S_ * DP), *kp_l = alloc16((size_t)B_ * HEADS * S_ * DP);
  u16 *vt_h = alloc16((size_t)B_ * HEADS * DVP * S_), *vt_l = alloc16((size_t)B_ * HEADS * DVP * S_);
  u16 *o_h  = alloc16((size_t)R_ * E_),        *o_l  = alloc16((size_t)R_ * E_);
  u16 *x2_h = alloc16((size_t)R_ * E_),        *x2_l = alloc16((size_t)R_ * E_);
  u16 *g_h  = alloc16((size_t)R_ * INTERP),    *g_l  = alloc16((size_t)R_ * INTERP);
  float* h1 = (float*)(base + off);
  off = (off + (size_t)R_ * E_ * sizeof(float) + 255) & ~(size_t)255;

  // weights -> bf16 hi/lo (L2-resident: 192MB L2 >> total weight bytes)
  k_conv_hilo<<<(3 * E_ * E_ + 255) / 256, 256, 0, stream>>>(qkvw, qkvw_h, qkvw_l, 3 * E_ * E_);
  k_conv_hilo<<<(E_ * E_ + 255) / 256, 256, 0, stream>>>(outw, outw_h, outw_l, E_ * E_);
  k_conv_hilo<<<(INTER * E_ + 255) / 256, 256, 0, stream>>>(fc1w, fc1w_h, fc1w_l, INTER * E_);
  k_conv_hilo_pad<<<(E_ * INTER + 255) / 256, 256, 0, stream>>>(fc2w, fc2w_h, fc2w_l, INTER, INTERP, E_ * INTER);

  // zero pads (u32 granularity)
  {
    int t1 = B_ * HEADS * S_ * 12;  // d pads [72,96) of qp/kp
    k_zero_pad<<<(t1 + 255) / 256, 256, 0, stream>>>((u32*)qp_h, 48, 36, 12, t1);
    k_zero_pad<<<(t1 + 255) / 256, 256, 0, stream>>>((u32*)qp_l, 48, 36, 12, t1);
    k_zero_pad<<<(t1 + 255) / 256, 256, 0, stream>>>((u32*)kp_h, 48, 36, 12, t1);
    k_zero_pad<<<(t1 + 255) / 256, 256, 0, stream>>>((u32*)kp_l, 48, 36, 12, t1);
    int t2 = B_ * HEADS * 4096;     // vT rows d in [72,80)
    k_zero_pad<<<(t2 + 255) / 256, 256, 0, stream>>>((u32*)vt_h, 40960, 36864, 4096, t2);
    k_zero_pad<<<(t2 + 255) / 256, 256, 0, stream>>>((u32*)vt_l, 40960, 36864, 4096, t2);
    int t3 = R_ * 8;                // G cols [4304,4320)
    k_zero_pad<<<(t3 + 255) / 256, 256, 0, stream>>>((u32*)g_h, 2160, 2152, 8, t3);
    k_zero_pad<<<(t3 + 255) / 256, 256, 0, stream>>>((u32*)g_l, 2160, 2152, 8, t3);
    int t4 = E_ * 8;                // fc2w cols [4304,4320)
    k_zero_pad<<<(t4 + 255) / 256, 256, 0, stream>>>((u32*)fc2w_h, 2160, 2152, 8, t4);
    k_zero_pad<<<(t4 + 255) / 256, 256, 0, stream>>>((u32*)fc2w_l, 2160, 2152, 8, t4);
  }

  // pipeline
  k_layernorm<<<R_ / 8, 256, 0, stream>>>(hs, ln1g, ln1b, xln_h, xln_l);
  k_gemm_qkv<<<dim3(R_ / 16, (3 * E_) / 64), 32, 0, stream>>>(
      xln_h, xln_l, qkvw_h, qkvw_l, qkvb, qp_h, qp_l, kp_h, kp_l, vt_h, vt_l);
  k_attn<<<B_ * HEADS * (S_ / 16), 32, 0, stream>>>(
      qp_h, qp_l, kp_h, kp_l, vt_h, vt_l, o_h, o_l);
  k_gemm_out<<<dim3(R_ / 16, E_ / 64), 32, 0, stream>>>(
      o_h, o_l, outw_h, outw_l, outb, hs, h1);
  k_layernorm<<<R_ / 8, 256, 0, stream>>>(h1, ln2g, ln2b, x2_h, x2_l);
  k_gemm_fc1<<<dim3(R_ / 16, 4288 / 64), 32, 0, stream>>>(
      x2_h, x2_l, fc1w_h, fc1w_l, fc1b, g_h, g_l);
  k_gemm_fc1_tail<<<R_ / 16, 32, 0, stream>>>(
      x2_h, x2_l, fc1w_h, fc1w_l, fc1b, g_h, g_l);
  k_gemm_fc2<<<dim3(R_ / 16, E_ / 64), 32, 0, stream>>>(
      g_h, g_l, fc2w_h, fc2w_l, fc2b, h1, out);
}
